// DistanceBasedClassifier_47579647705097
// MI455X (gfx1250) — matrix operand order repaired and checked by reading, not compiled, not run on hardware
//
#include <hip/hip_runtime.h>
#include <hip/hip_bf16.h>
#include <math.h>

// ---------------------------------------------------------------------------
// min_j ||x_i - y_j|| * 10  via split-bf16 WMMA GEMM with fused min epilogue.
//   x (mutation_dist): [8192, 96] f32     y (train_data): [65536, 96] f32
// Compute-bound (103 GFLOP). f32 = bf16_hi + bf16_lo; x.y ~= hi.hi + hi.lo + lo.hi
// (3x v_wmma_f32_16x16x32_bf16). min_j dist^2 = x2 + min_j (y2 - 2xy); sqrt at end.
// A fragments live in registers for the whole block; B tiles are double-buffered
// in LDS via GLOBAL_LOAD_ASYNC_TO_LDS_B128 (ASYNCcnt) for copy/compute overlap.
// ---------------------------------------------------------------------------

typedef __attribute__((ext_vector_type(16))) __bf16 v16bf;
typedef __attribute__((ext_vector_type(8)))  float  v8f;

#define N_ROWS 8192
#define M_COLS 65536
#define KDIM   96
#define BM     128
#define BN     128
#define LDSP   104          // padded LDS K-stride (elements): 208B -> 16 distinct banks
#define CT_PER_BLOCK 32     // column tiles (of BN) per block
#define COL_SPLITS   16     // gridDim.y

__device__ __forceinline__ unsigned short f2bf_rne(float f) {
    unsigned u = __float_as_uint(f);
    unsigned r = 0x7FFFu + ((u >> 16) & 1u);
    return (unsigned short)((u + r) >> 16);
}
__device__ __forceinline__ float bf2f(unsigned short h) {
    return __uint_as_float(((unsigned)h) << 16);
}
// monotonic float -> u32 key (handles negatives) for atomicMin
__device__ __forceinline__ unsigned fkey(float f) {
    unsigned u = __float_as_uint(f);
    return (u & 0x80000000u) ? ~u : (u | 0x80000000u);
}
__device__ __forceinline__ float funkey(unsigned k) {
    return (k & 0x80000000u) ? __uint_as_float(k & 0x7FFFFFFFu) : __uint_as_float(~k);
}

// ---------------------------------------------------------------------------
__global__ void init_min_kernel(unsigned* __restrict__ acc, int n) {
    int i = blockIdx.x * blockDim.x + threadIdx.x;
    if (i < n) acc[i] = 0xFFFFFFFFu;
}

// Split rows of src [nrows,96] into bf16 hi/lo and compute per-row sum of squares.
__global__ void prep_kernel(const float* __restrict__ src,
                            unsigned short* __restrict__ hi,
                            unsigned short* __restrict__ lo,
                            float* __restrict__ sumsq, int nrows) {
    const int lane = threadIdx.x & 31;
    const int wave = threadIdx.x >> 5;
    const int row  = blockIdx.x * 8 + wave;
    if (row >= nrows) return;
    float s = 0.0f;
#pragma unroll
    for (int i = 0; i < 3; ++i) {
        int k = lane + i * 32;
        float v = src[row * KDIM + k];
        unsigned short h = f2bf_rne(v);
        float hf = bf2f(h);
        unsigned short l = f2bf_rne(v - hf);
        hi[row * KDIM + k] = h;
        lo[row * KDIM + k] = l;
        s = fmaf(v, v, s);
    }
#pragma unroll
    for (int off = 16; off >= 1; off >>= 1) s += __shfl_xor(s, off);
    if (lane == 0) sumsq[row] = s;
}

// ---------------------------------------------------------------------------
union Frag { uint4 q[2]; v16bf v; };

// Issue one B tile (hi+lo) -> LDS buffer via async-to-LDS DMA (12 x b128 / thread).
__device__ __forceinline__ void prefetch_tile(const unsigned short* __restrict__ Bh,
                                              const unsigned short* __restrict__ Bl,
                                              unsigned ldsH, unsigned ldsL,
                                              int colBase, int tid) {
#pragma unroll
    for (int i = 0; i < 6; ++i) {
        int id = tid + i * 256;              // 1536 chunks of 8 bf16 per array
        int r  = id / 12;                    // 12 chunks per 96-elem row
        int c  = id % 12;
        unsigned dh = ldsH + (unsigned)(r * 13 + c) * 16u;   // LDS row = 13 uint4 (104 elems)
        unsigned dl = ldsL + (unsigned)(r * 13 + c) * 16u;
        unsigned long long gh =
            (unsigned long long)(uintptr_t)(Bh + (size_t)(colBase + r) * KDIM + c * 8);
        unsigned long long gl =
            (unsigned long long)(uintptr_t)(Bl + (size_t)(colBase + r) * KDIM + c * 8);
        asm volatile("global_load_async_to_lds_b128 %0, %1, off" :: "v"(dh), "v"(gh) : "memory");
        asm volatile("global_load_async_to_lds_b128 %0, %1, off" :: "v"(dl), "v"(gl) : "memory");
    }
}

// Main kernel: block tile 128x128, 8 waves; wave (wr,wc) owns 32x64 of C.
__global__ __launch_bounds__(256)
void cdist_min_wmma_kernel(const unsigned short* __restrict__ Ah,
                           const unsigned short* __restrict__ Al,
                           const unsigned short* __restrict__ Bh,
                           const unsigned short* __restrict__ Bl,
                           const float* __restrict__ y2,
                           unsigned* __restrict__ minacc) {
    __shared__ __attribute__((aligned(16))) unsigned short sBh[2][BN * LDSP];
    __shared__ __attribute__((aligned(16))) unsigned short sBl[2][BN * LDSP];

    const int tid  = threadIdx.x;
    const int lane = tid & 31;
    const int wave = tid >> 5;
    const int wr   = wave & 3;        // 4 row sub-tiles of 32
    const int wc   = wave >> 2;       // 2 col sub-tiles of 64
    const int hh   = lane >> 4;       // half-wave id (0/1)
    const int l15  = lane & 15;

    const int rowBase  = blockIdx.x * BM;
    const int colBase0 = blockIdx.y * (CT_PER_BLOCK * BN);

    const unsigned ldsH0 = (unsigned)(uintptr_t)&sBh[0][0];
    const unsigned ldsH1 = (unsigned)(uintptr_t)&sBh[1][0];
    const unsigned ldsL0 = (unsigned)(uintptr_t)&sBl[0][0];
    const unsigned ldsL1 = (unsigned)(uintptr_t)&sBl[1][0];

    // ---- kick off async prefetch of B tile 0 into buffer 0 ----
    prefetch_tile(Bh, Bl, ldsH0, ldsL0, colBase0, tid);

    // ---- hoist all A fragments (hi+lo, 3 k-chunks, 2 m-tiles) into registers ----
    // A 16x32 bf16 fragment: lane row = l15, K runs [kb,kb+8) & [kb+16,kb+24), kb=(lane>>4)*8
    Frag AH[3][2], AL[3][2];
#pragma unroll
    for (int kc = 0; kc < 3; ++kc) {
#pragma unroll
        for (int m = 0; m < 2; ++m) {
            int row = rowBase + wr * 32 + m * 16 + l15;
            int kb  = kc * 32 + hh * 8;
            const unsigned short* ph = Ah + (size_t)row * KDIM + kb;
            const unsigned short* pl = Al + (size_t)row * KDIM + kb;
            AH[kc][m].q[0] = *(const uint4*)(ph);
            AH[kc][m].q[1] = *(const uint4*)(ph + 16);
            AL[kc][m].q[0] = *(const uint4*)(pl);
            AL[kc][m].q[1] = *(const uint4*)(pl + 16);
        }
    }

    float rowmin[2][8];
#pragma unroll
    for (int m = 0; m < 2; ++m)
#pragma unroll
        for (int r = 0; r < 8; ++r) rowmin[m][r] = __builtin_inff();

    for (int t = 0; t < CT_PER_BLOCK; ++t) {
        const int cur = t & 1;
        const int colBase = colBase0 + t * BN;

        if (t + 1 < CT_PER_BLOCK) {
            // prefetch next tile into the other buffer (freed by last iteration's barrier)
            prefetch_tile(Bh, Bl, cur ? ldsH0 : ldsH1, cur ? ldsL0 : ldsL1,
                          colBase + BN, tid);
            asm volatile("s_wait_asynccnt 12" ::: "memory");   // tile t resident; prefetch in flight
        } else {
            asm volatile("s_wait_asynccnt 0" ::: "memory");
        }
        __syncthreads();

        const unsigned short* sH = sBh[cur];
        const unsigned short* sL = sBl[cur];

        v8f acc[2][4];
#pragma unroll
        for (int m = 0; m < 2; ++m)
#pragma unroll
            for (int n = 0; n < 4; ++n) acc[m][n] = (v8f){};

#pragma unroll
        for (int kc = 0; kc < 3; ++kc) {
            const int k0 = kc * 32;
#pragma unroll
            for (int n = 0; n < 4; ++n) {
                // B 32x16 bf16 fragment: lane col = l15, K runs [kb,kb+8) & [kb+16,kb+24), kb=(lane>>4)*8
                int brow = wc * 64 + n * 16 + l15;
                int kb   = k0 + hh * 8;
                Frag bh, bl;
                const unsigned short* ph = &sH[brow * LDSP + kb];
                const unsigned short* pl = &sL[brow * LDSP + kb];
                bh.q[0] = *(const uint4*)(ph);
                bh.q[1] = *(const uint4*)(ph + 16);
                bl.q[0] = *(const uint4*)(pl);
                bl.q[1] = *(const uint4*)(pl + 16);
#pragma unroll
                for (int m = 0; m < 2; ++m) {
                    acc[m][n] = __builtin_amdgcn_wmma_f32_16x16x32_bf16(
                        false, AH[kc][m].v, false, bh.v, (short)0, acc[m][n], false, false);
                    acc[m][n] = __builtin_amdgcn_wmma_f32_16x16x32_bf16(
                        false, AH[kc][m].v, false, bl.v, (short)0, acc[m][n], false, false);
                    acc[m][n] = __builtin_amdgcn_wmma_f32_16x16x32_bf16(
                        false, AL[kc][m].v, false, bh.v, (short)0, acc[m][n], false, false);
                }
            }
        }

        // ---- fused epilogue: rowmin over (y2 - 2*xy); 1 fma + 1 min per element ----
#pragma unroll
        for (int n = 0; n < 4; ++n) {
            float y2v = y2[colBase + wc * 64 + n * 16 + l15];
#pragma unroll
            for (int m = 0; m < 2; ++m)
#pragma unroll
                for (int r = 0; r < 8; ++r)
                    rowmin[m][r] = fminf(rowmin[m][r], fmaf(acc[m][n][r], -2.0f, y2v));
        }
        __syncthreads();   // everyone done reading buf[cur] before it is re-prefetched
    }

    // cross-lane min within each 16-lane half (same C row), then global atomic
#pragma unroll
    for (int m = 0; m < 2; ++m) {
#pragma unroll
        for (int r = 0; r < 8; ++r) {
            float v = rowmin[m][r];
            v = fminf(v, __shfl_xor(v, 1));
            v = fminf(v, __shfl_xor(v, 2));
            v = fminf(v, __shfl_xor(v, 4));
            v = fminf(v, __shfl_xor(v, 8));
            if (l15 == 0) {
                int row = rowBase + wr * 32 + m * 16 + hh * 8 + r;
                atomicMin(&minacc[row], fkey(v));
            }
        }
    }
}

__global__ void finish_kernel(const unsigned* __restrict__ acc,
                              const float* __restrict__ x2,
                              float* __restrict__ out, int n) {
    int i = blockIdx.x * blockDim.x + threadIdx.x;
    if (i < n) {
        float v = x2[i] + funkey(acc[i]);
        out[i] = 10.0f * sqrtf(fmaxf(v, 0.0f));
    }
}

// ---------------------------------------------------------------------------
extern "C" void kernel_launch(void* const* d_in, const int* in_sizes, int n_in,
                              void* d_out, int out_size, void* d_ws, size_t ws_size,
                              hipStream_t stream) {
    const float* x = (const float*)d_in[0];   // mutation_dist [8192,96]
    const float* y = (const float*)d_in[1];   // train_data  [65536,96]
    float* out = (float*)d_out;               // [8192]

    // workspace carve (all sizes 16B multiples)
    char* p = (char*)d_ws;
    unsigned short* Ah = (unsigned short*)p; p += (size_t)N_ROWS * KDIM * 2;
    unsigned short* Al = (unsigned short*)p; p += (size_t)N_ROWS * KDIM * 2;
    unsigned short* Bh = (unsigned short*)p; p += (size_t)M_COLS * KDIM * 2;
    unsigned short* Bl = (unsigned short*)p; p += (size_t)M_COLS * KDIM * 2;
    float* x2 = (float*)p;       p += (size_t)N_ROWS * 4;
    float* y2 = (float*)p;       p += (size_t)M_COLS * 4;
    unsigned* minacc = (unsigned*)p;

    init_min_kernel<<<(N_ROWS + 255) / 256, 256, 0, stream>>>(minacc, N_ROWS);
    prep_kernel<<<N_ROWS / 8, 256, 0, stream>>>(x, Ah, Al, x2, N_ROWS);
    prep_kernel<<<M_COLS / 8, 256, 0, stream>>>(y, Bh, Bl, y2, M_COLS);

    dim3 grid(N_ROWS / BM, COL_SPLITS);
    cdist_min_wmma_kernel<<<grid, 256, 0, stream>>>(Ah, Al, Bh, Bl, y2, minacc);

    finish_kernel<<<(N_ROWS + 255) / 256, 256, 0, stream>>>(minacc, x2, out, N_ROWS);
}
